// GATNet_62423054680285
// MI455X (gfx1250) — matrix operand order repaired and checked by reading, not compiled, not run on hardware
//
#include <hip/hip_runtime.h>
#include <hip/hip_fp16.h>

#define NEG_SLOPE 0.2f

typedef __attribute__((ext_vector_type(16))) _Float16 v16h;
typedef __attribute__((ext_vector_type(8)))  _Float16 v8h;
typedef __attribute__((ext_vector_type(8)))  float    v8f;
typedef __attribute__((ext_vector_type(4)))  unsigned v4u;
typedef __attribute__((ext_vector_type(8)))  int      v8i;
typedef __attribute__((ext_vector_type(4)))  int      v4i;

// ---------------- helpers ----------------
__device__ __forceinline__ unsigned fenc(float f) {
    unsigned u = __float_as_uint(f);
    return (u & 0x80000000u) ? ~u : (u | 0x80000000u);
}
__device__ __forceinline__ float fdec(unsigned u) {
    unsigned b = (u & 0x80000000u) ? (u & 0x7fffffffu) : ~u;
    return __uint_as_float(b);
}
__device__ __forceinline__ float lrelu(float x) { return x > 0.f ? x : NEG_SLOPE * x; }

__device__ __forceinline__ unsigned to_lds_addr(const void* p) {
    return (unsigned)(unsigned long long)(const __attribute__((address_space(3))) char*)p;
}

#if __has_builtin(__builtin_amdgcn_tensor_load_to_lds)
// Linear DMA: nelem f16 elements from global -> LDS via the Tensor Data Mover.
// D# per CDNA5 ISA: 1-row 2D tensor, tile == tensor.
// 6-arg builtin form: (v4u g0, v8i g1, v4i g2, v4i g3, v8i g4, i32 cpol)
__device__ __forceinline__ void tdm_copy_f16(unsigned lds_addr, const void* gptr, unsigned nelem) {
    unsigned long long ga = (unsigned long long)gptr;
    v4u g0;
    g0.x = 1u;                                   // count=1 (valid user descriptor)
    g0.y = lds_addr;                             // lds_addr (bytes)
    g0.z = (unsigned)(ga & 0xffffffffu);         // global_addr[31:0]
    g0.w = (unsigned)((ga >> 32) & 0x01ffffffu)  // global_addr[56:32]
         | (2u << 30);                           // type = 2 ("image")
    v8i g1;
    g1[0] = (int)(1u << 16);                         // data_size = 1 -> 2 bytes
    g1[1] = (int)((nelem & 0xffffu) << 16);          // tensor_dim0[15:0]  @ bits[63:48]
    g1[2] = (int)((nelem >> 16) & 0xffffu)           // tensor_dim0[31:16] @ bits[79:64]
          | (int)(1u << 16);                         // tensor_dim1 = 1    @ bits[95:80]
    g1[3] = (int)((nelem & 0xffffu) << 16);          // tile_dim0 @ bits[127:112]
    g1[4] = 1;                                       // tile_dim1 = 1
    g1[5] = (int)nelem;                              // tensor_dim0_stride[31:0]
    g1[6] = 0;
    g1[7] = 0;
    v4i z4 = {0, 0, 0, 0};
    v8i z8 = {0, 0, 0, 0, 0, 0, 0, 0};
    __builtin_amdgcn_tensor_load_to_lds(g0, g1, z4, z4, z8, 0);
}
#endif

// ---------------- elementwise f32 -> f16 ----------------
__global__ void k_cvt_f16(const float* __restrict__ in, _Float16* __restrict__ out, int n) {
    int i = blockIdx.x * blockDim.x + threadIdx.x;
    if (i < n) out[i] = (_Float16)in[i];
}

// ---------------- build transposed f16 weights: Wt[n][k] = W[k][n] ----------------
__global__ void k_prep_wt(const float* __restrict__ W, _Float16* __restrict__ Wt,
                          int K, int Ncols, int Npad) {
    int i = blockIdx.x * blockDim.x + threadIdx.x;
    if (i >= Npad * K) return;
    int n = i / K, k = i - n * K;
    Wt[i] = (n < Ncols) ? (_Float16)W[(size_t)k * Ncols + n] : (_Float16)0.f;
}

// ---------------- GEMM1: H[N,64] = Xh[N,256] @ W1[256,64]  (256 rows/block) --------
__global__ __launch_bounds__(128) void k_gemm1(const _Float16* __restrict__ Wt_g,
                                               const _Float16* __restrict__ X,
                                               float* __restrict__ H, int N) {
    __shared__ _Float16 Wt[64 * 256];  // 32 KB, W1^T in f16
#if __has_builtin(__builtin_amdgcn_tensor_load_to_lds)
    if (threadIdx.x < 32) {
        tdm_copy_f16(to_lds_addr(Wt), Wt_g, 64 * 256);
        __builtin_amdgcn_s_wait_tensorcnt(0);
    }
#else
    for (int i = threadIdx.x; i < 64 * 256; i += 128) Wt[i] = Wt_g[i];
#endif
    __syncthreads();

    int wave = threadIdx.x >> 5, lane = threadIdx.x & 31;
    int m = lane & 15, hi = lane >> 4;

    for (int t = 0; t < 4; ++t) {
        int row0 = blockIdx.x * 256 + t * 64 + wave * 16;
        if (row0 >= N) break;
        int r = row0 + m; if (r > N - 1) r = N - 1;      // clamp OOB loads
        const _Float16* xrow = X + (size_t)r * 256;

        v8f acc[4] = {};
#pragma unroll
        for (int k0 = 0; k0 < 256; k0 += 32) {
            const _Float16* pa = xrow + k0 + hi * 16;
            __builtin_prefetch(pa + 64, 0, 0);           // global_prefetch_b8
            v8h alo = *(const v8h*)(pa);
            v8h ahi = *(const v8h*)(pa + 8);
            v16h a = __builtin_shufflevector(alo, ahi, 0,1,2,3,4,5,6,7,8,9,10,11,12,13,14,15);
#pragma unroll
            for (int nt = 0; nt < 4; ++nt) {
                v16h b = *(const v16h*)&Wt[(nt * 16 + m) * 256 + k0 + hi * 16];
                acc[nt] = __builtin_amdgcn_wmma_f32_16x16x32_f16(
                    false, a, false, b, (short)0, acc[nt], false, false);
            }
        }
#pragma unroll
        for (int nt = 0; nt < 4; ++nt) {
            int col = nt * 16 + m;
#pragma unroll
            for (int rr = 0; rr < 8; ++rr) {
                int row = row0 + rr + hi * 8;
                if (row < N) H[(size_t)row * 64 + col] = acc[nt][rr];
            }
        }
    }
}

// ---------------- GEMM2: H2[N,40] = Hh[N,64] @ W2[64,40]  (256 rows/block) ---------
__global__ __launch_bounds__(128) void k_gemm2(const _Float16* __restrict__ Wt_g,
                                               const _Float16* __restrict__ X,
                                               float* __restrict__ H, int N) {
    __shared__ _Float16 Wt[48 * 64];   // 6 KB, W2^T padded to 48 cols
#if __has_builtin(__builtin_amdgcn_tensor_load_to_lds)
    if (threadIdx.x < 32) {
        tdm_copy_f16(to_lds_addr(Wt), Wt_g, 48 * 64);
        __builtin_amdgcn_s_wait_tensorcnt(0);
    }
#else
    for (int i = threadIdx.x; i < 48 * 64; i += 128) Wt[i] = Wt_g[i];
#endif
    __syncthreads();

    int wave = threadIdx.x >> 5, lane = threadIdx.x & 31;
    int m = lane & 15, hi = lane >> 4;

    for (int t = 0; t < 4; ++t) {
        int row0 = blockIdx.x * 256 + t * 64 + wave * 16;
        if (row0 >= N) break;
        int r = row0 + m; if (r > N - 1) r = N - 1;
        const _Float16* xrow = X + (size_t)r * 64;

        v8f acc[3] = {};
#pragma unroll
        for (int k0 = 0; k0 < 64; k0 += 32) {
            const _Float16* pa = xrow + k0 + hi * 16;
            v8h alo = *(const v8h*)(pa);
            v8h ahi = *(const v8h*)(pa + 8);
            v16h a = __builtin_shufflevector(alo, ahi, 0,1,2,3,4,5,6,7,8,9,10,11,12,13,14,15);
#pragma unroll
            for (int nt = 0; nt < 3; ++nt) {
                v16h b = *(const v16h*)&Wt[(nt * 16 + m) * 64 + k0 + hi * 16];
                acc[nt] = __builtin_amdgcn_wmma_f32_16x16x32_f16(
                    false, a, false, b, (short)0, acc[nt], false, false);
            }
        }
#pragma unroll
        for (int nt = 0; nt < 3; ++nt) {
            int col = nt * 16 + m;
            if (col >= 40) continue;
#pragma unroll
            for (int rr = 0; rr < 8; ++rr) {
                int row = row0 + rr + hi * 8;
                if (row < N) H[(size_t)row * 40 + col] = acc[nt][rr];
            }
        }
    }
}

// ---------------- per-node attention logits: [N,H] ----------------
__global__ void k_att(const float* __restrict__ h, const float* __restrict__ as_,
                      const float* __restrict__ ad_, float* __restrict__ ols,
                      float* __restrict__ old_, int N, int H, int C) {
    int t = blockIdx.x * blockDim.x + threadIdx.x;
    if (t >= N * H) return;
    int n = t / H, hh = t - n * H;
    const float* hp = h + (size_t)n * H * C + (size_t)hh * C;
    const float* ap = as_ + hh * C;
    const float* bp = ad_ + hh * C;
    float s = 0.f, d = 0.f;
    for (int c = 0; c < C; ++c) { float v = hp[c]; s += v * ap[c]; d += v * bp[c]; }
    ols[t] = s; old_[t] = d;
}

// ---------------- edge pass 1: segment max (ordered-uint atomicMax) --------
__global__ void k_edge_max(const int* __restrict__ ei, int E, int N, int H,
                           const float* __restrict__ als, const float* __restrict__ ald,
                           unsigned* __restrict__ menc) {
    int e = blockIdx.x * blockDim.x + threadIdx.x;
    if (e >= E + N) return;
    int s, d;
    if (e < E) { s = ei[e]; d = ei[E + e]; } else { s = e - E; d = s; }
    for (int hh = 0; hh < H; ++hh) {
        float v = lrelu(als[s * H + hh] + ald[d * H + hh]);
        atomicMax(&menc[d * H + hh], fenc(v));
    }
}

// ---------------- edge pass 2: denom = segment_sum exp(e - m) --------------
__global__ void k_edge_den(const int* __restrict__ ei, int E, int N, int H,
                           const float* __restrict__ als, const float* __restrict__ ald,
                           const unsigned* __restrict__ menc, float* __restrict__ den) {
    int e = blockIdx.x * blockDim.x + threadIdx.x;
    if (e >= E + N) return;
    int s, d;
    if (e < E) { s = ei[e]; d = ei[E + e]; } else { s = e - E; d = s; }
    for (int hh = 0; hh < H; ++hh) {
        float v = lrelu(als[s * H + hh] + ald[d * H + hh]);
        atomicAdd(&den[d * H + hh], __expf(v - fdec(menc[d * H + hh])));
    }
}

// ---------------- edge pass 3: out[d] += alpha * feat[s] -------------------
__global__ void k_edge_agg(const int* __restrict__ ei, int E, int N, int H, int C,
                           const float* __restrict__ als, const float* __restrict__ ald,
                           const unsigned* __restrict__ menc, const float* __restrict__ den,
                           const float* __restrict__ feat, float* __restrict__ out) {
    long long t = (long long)blockIdx.x * blockDim.x + threadIdx.x;
    if (t >= (long long)(E + N) * H) return;
    int e = (int)(t / H), hh = (int)(t - (long long)(t / H) * H);
    int s, d;
    if (e < E) { s = ei[e]; d = ei[E + e]; } else { s = e - E; d = s; }
    float v = lrelu(als[s * H + hh] + ald[d * H + hh]);
    float alpha = __expf(v - fdec(menc[d * H + hh])) / den[d * H + hh];
    const float* fp = feat + (size_t)s * H * C + (size_t)hh * C;
    float* op = out + (size_t)d * H * C + (size_t)hh * C;
    for (int c = 0; c < C; ++c) atomicAdd(&op[c], alpha * fp[c]);
}

// ---------------- conv1 finalize: +bias, ELU, cast f16 ----------------
__global__ void k_fin1(const float* __restrict__ agg, const float* __restrict__ b,
                       _Float16* __restrict__ hh, int N) {
    int t = blockIdx.x * blockDim.x + threadIdx.x;
    if (t >= N * 64) return;
    int j = t & 63;
    float v = agg[t] + b[j];
    v = v > 0.f ? v : (__expf(v) - 1.f);
    hh[t] = (_Float16)v;
}

// ---------------- conv2 finalize: +bias, log_softmax over 40 ----------------
__global__ void k_fin2(const float* __restrict__ agg, const float* __restrict__ b,
                       float* __restrict__ y, int N) {
    int n = blockIdx.x * blockDim.x + threadIdx.x;
    if (n >= N) return;
    float v[40]; float mx = -1e30f;
    for (int c = 0; c < 40; ++c) { v[c] = agg[n * 40 + c] + b[c]; mx = fmaxf(mx, v[c]); }
    float sum = 0.f;
    for (int c = 0; c < 40; ++c) sum += __expf(v[c] - mx);
    float l = __logf(sum) + mx;
    for (int c = 0; c < 40; ++c) y[(size_t)n * 40 + c] = v[c] - l;
}

// ---------------- launch ----------------
extern "C" void kernel_launch(void* const* d_in, const int* in_sizes, int n_in,
                              void* d_out, int out_size, void* d_ws, size_t ws_size,
                              hipStream_t stream) {
    const float* x    = (const float*)d_in[0];
    const int*   ei   = (const int*)d_in[1];
    const float* W1   = (const float*)d_in[2];
    const float* as1  = (const float*)d_in[3];
    const float* ad1  = (const float*)d_in[4];
    const float* b1   = (const float*)d_in[5];
    const float* W2   = (const float*)d_in[6];
    const float* as2  = (const float*)d_in[7];
    const float* ad2  = (const float*)d_in[8];
    const float* b2   = (const float*)d_in[9];
    float* y = (float*)d_out;

    const int N  = in_sizes[0] / 256;
    const int E  = in_sizes[1] / 2;
    const int ET = E + N;

    char* base = (char*)d_ws;
    size_t off = 0;
    auto alloc = [&](size_t bytes) { size_t o = off; off = (off + bytes + 255) & ~(size_t)255; return o; };
    size_t o_xh   = alloc((size_t)N * 256 * 2);
    size_t o_wt1  = alloc((size_t)64 * 256 * 2);
    size_t o_wt2  = alloc((size_t)48 * 64 * 2);
    size_t o_h1   = alloc((size_t)N * 64 * 4);
    size_t o_al1s = alloc((size_t)N * 8 * 4);
    size_t o_al1d = alloc((size_t)N * 8 * 4);
    size_t o_m1   = alloc((size_t)N * 8 * 4);
    size_t o_den1 = alloc((size_t)N * 8 * 4);
    size_t o_out1 = alloc((size_t)N * 64 * 4);
    size_t o_h1h  = alloc((size_t)N * 64 * 2);
    size_t o_h2   = alloc((size_t)N * 40 * 4);
    size_t o_al2s = alloc((size_t)N * 4);
    size_t o_al2d = alloc((size_t)N * 4);
    size_t o_m2   = alloc((size_t)N * 4);
    size_t o_den2 = alloc((size_t)N * 4);
    size_t o_out2 = alloc((size_t)N * 40 * 4);

    _Float16* xh   = (_Float16*)(base + o_xh);
    _Float16* wt1  = (_Float16*)(base + o_wt1);
    _Float16* wt2  = (_Float16*)(base + o_wt2);
    float*    h1   = (float*)(base + o_h1);
    float*    al1s = (float*)(base + o_al1s);
    float*    al1d = (float*)(base + o_al1d);
    unsigned* m1   = (unsigned*)(base + o_m1);
    float*    den1 = (float*)(base + o_den1);
    float*    out1 = (float*)(base + o_out1);
    _Float16* h1h  = (_Float16*)(base + o_h1h);
    float*    h2   = (float*)(base + o_h2);
    float*    al2s = (float*)(base + o_al2s);
    float*    al2d = (float*)(base + o_al2d);
    unsigned* m2   = (unsigned*)(base + o_m2);
    float*    den2 = (float*)(base + o_den2);
    float*    out2 = (float*)(base + o_out2);

    // ---- weight prep (once per call, tiny) ----
    k_prep_wt<<<(64 * 256 + 255) / 256, 256, 0, stream>>>(W1, wt1, 256, 64, 64);
    k_prep_wt<<<(48 * 64 + 255) / 256, 256, 0, stream>>>(W2, wt2, 64, 40, 48);

    // ---- layer 1 ----
    k_cvt_f16<<<(N * 256 + 255) / 256, 256, 0, stream>>>(x, xh, N * 256);
    k_gemm1<<<(N + 255) / 256, 128, 0, stream>>>(wt1, xh, h1, N);
    k_att<<<(N * 8 + 255) / 256, 256, 0, stream>>>(h1, as1, ad1, al1s, al1d, N, 8, 8);
    (void)hipMemsetAsync(m1,   0, (size_t)N * 8 * 4, stream);
    (void)hipMemsetAsync(den1, 0, (size_t)N * 8 * 4, stream);
    (void)hipMemsetAsync(out1, 0, (size_t)N * 64 * 4, stream);
    k_edge_max<<<(ET + 255) / 256, 256, 0, stream>>>(ei, E, N, 8, al1s, al1d, m1);
    k_edge_den<<<(ET + 255) / 256, 256, 0, stream>>>(ei, E, N, 8, al1s, al1d, m1, den1);
    {
        long long tot = (long long)ET * 8;
        k_edge_agg<<<(unsigned)((tot + 255) / 256), 256, 0, stream>>>(
            ei, E, N, 8, 8, al1s, al1d, m1, den1, h1, out1);
    }
    k_fin1<<<(N * 64 + 255) / 256, 256, 0, stream>>>(out1, b1, h1h, N);

    // ---- layer 2 ----
    k_gemm2<<<(N + 255) / 256, 128, 0, stream>>>(wt2, h1h, h2, N);
    k_att<<<(N + 255) / 256, 256, 0, stream>>>(h2, as2, ad2, al2s, al2d, N, 1, 40);
    (void)hipMemsetAsync(m2,   0, (size_t)N * 4, stream);
    (void)hipMemsetAsync(den2, 0, (size_t)N * 4, stream);
    (void)hipMemsetAsync(out2, 0, (size_t)N * 40 * 4, stream);
    k_edge_max<<<(ET + 255) / 256, 256, 0, stream>>>(ei, E, N, 1, al2s, al2d, m2);
    k_edge_den<<<(ET + 255) / 256, 256, 0, stream>>>(ei, E, N, 1, al2s, al2d, m2, den2);
    k_edge_agg<<<(ET + 255) / 256, 256, 0, stream>>>(ei, E, N, 1, 40, al2s, al2d, m2, den2, h2, out2);
    k_fin2<<<(N + 127) / 128, 128, 0, stream>>>(out2, b2, y, N);
}
